// KWS_LSTM_bmm_24163486007740
// MI455X (gfx1250) — 8-bit matrix operand order repaired and checked by reading, compiled, not run on hardware
//
#include <hip/hip_runtime.h>
#include <hip/hip_bf16.h>


#define T_STEPS 100
#define BATCH   1024
#define INDIM   40
#define HDIM    512
#define NBLK    4
#define GATES   (4 * HDIM)
#define KT_HH   (HDIM / 64)

typedef int v8i __attribute__((ext_vector_type(8)));

__device__ __forceinline__ float clamp_pm(float v, float a) {
    return fminf(fmaxf(v, -a), a);
}
__device__ __forceinline__ float pact_q8(float v, float a) {
    float t = rintf(fminf(fmaxf(v * (128.f / a), -127.f), 127.f));
    return t * (a * 0.0078125f);
}
__device__ __forceinline__ float quant8_r1(float v) {
    return rintf(fminf(fmaxf(v * 128.f, -127.f), 127.f)) * 0.0078125f;
}
__device__ __forceinline__ float sigmoidf(float x) {
    return 1.f / (1.f + expf(-x));
}

__device__ __forceinline__ v8i load_a8(const signed char* p) {
    v8i a;
#pragma unroll
    for (int i = 0; i < 4; ++i) {
        int2 t = *(const int2*)(p + i * 16);
        a[2 * i] = t.x;
        a[2 * i + 1] = t.y;
    }
    return a;
}
__device__ __forceinline__ v8i load_b8(const signed char* p) {
    v8i b;
    int4 t0 = *(const int4*)(p);
    int4 t1 = *(const int4*)(p + 16);
    b[0] = t0.x; b[1] = t0.y; b[2] = t0.z; b[3] = t0.w;
    b[4] = t1.x; b[5] = t1.y; b[6] = t1.z; b[7] = t1.w;
    return b;
}

__device__ __forceinline__ int perm_k(int pos) {
    return ((pos >> 3) & 3) * 16 + ((pos & 32) ? 8 : 0) + (pos & 7);
}

__global__ __launch_bounds__(256) void pack_whh_kernel(
    const float* __restrict__ w, signed char* __restrict__ outp) {
    size_t i = (size_t)blockIdx.x * 256 + threadIdx.x;
    if (i >= (size_t)NBLK * KT_HH * GATES * 64) return;
    int pos = (int)(i & 63);
    size_t t = i >> 6;
    int n = (int)(t % GATES); t /= GATES;
    int kt = (int)(t % KT_HH);
    int nb = (int)(t / KT_HH);
    int k = kt * 64 + perm_k(pos);
    float v = w[((size_t)nb * HDIM + k) * GATES + n];
    outp[i] = (signed char)rintf(fminf(fmaxf(v * 128.f, -127.f), 127.f));
}

__global__ __launch_bounds__(256) void pack_wih_kernel(
    const float* __restrict__ w, signed char* __restrict__ outp) {
    size_t i = (size_t)blockIdx.x * 256 + threadIdx.x;
    if (i >= (size_t)NBLK * GATES * 64) return;
    int pos = (int)(i & 63);
    size_t t = i >> 6;
    int n = (int)(t % GATES);
    int nb = (int)(t / GATES);
    int k = perm_k(pos);
    float v = (k < INDIM) ? w[((size_t)nb * INDIM + k) * GATES + n] : 0.f;
    outp[i] = (signed char)rintf(fminf(fmaxf(v * 128.f, -127.f), 127.f));
}

__global__ __launch_bounds__(256) void zero_state_kernel(
    signed char* __restrict__ h0, float* __restrict__ c) {
    size_t i = (size_t)blockIdx.x * 256 + threadIdx.x;
    if (i >= (size_t)NBLK * BATCH * HDIM) return;
    h0[i] = 0;
    c[i] = 0.f;
}

__global__ __launch_bounds__(256) void lstm_step_kernel(
    const float* __restrict__ x_t,
    const signed char* __restrict__ h_in,
    signed char* __restrict__ h_out,
    float* __restrict__ c,
    const signed char* __restrict__ wih_p,
    const signed char* __restrict__ whh_p,
    const float* __restrict__ bih,
    const float* __restrict__ bhh,
    const float* __restrict__ a_cell) {
    const int hc0 = blockIdx.x * 16;
    const int r0 = blockIdx.y * 256;
    const int nb = blockIdx.z;
    const int tid = threadIdx.x;
    const int wave = tid >> 5;
    const int lane = tid & 31;
    const int hi = lane >> 4;
    const int lm = lane & 15;

    const float a1  = a_cell[0 * NBLK + nb];
    const float a3  = a_cell[2 * NBLK + nb];
    const float a4  = a_cell[3 * NBLK + nb];
    const float a5  = a_cell[4 * NBLK + nb];
    const float a6  = a_cell[5 * NBLK + nb];
    const float a7  = a_cell[6 * NBLK + nb];
    const float a8  = a_cell[7 * NBLK + nb];
    const float a9  = a_cell[8 * NBLK + nb];
    const float a10 = a_cell[9 * NBLK + nb];
    const float a11 = a_cell[10 * NBLK + nb];
    const float a12 = a_cell[11 * NBLK + nb];
    const float a13 = a_cell[12 * NBLK + nb];
    const float a14 = a_cell[13 * NBLK + nb];

    __shared__ signed char xq[256 * 64];
    for (int i = tid; i < 256 * 64; i += 256) {
        int r = i >> 6, k = i & 63;
        float v = (k < INDIM) ? x_t[(size_t)(r0 + r) * INDIM + k] : 0.f;
        xq[i] = (signed char)rintf(fminf(fmaxf(v * (128.f / a1), -127.f), 127.f));
    }
    __syncthreads();

    v8i accX[2][4] = {{{0,0,0,0,0,0,0,0}, {0,0,0,0,0,0,0,0},
                       {0,0,0,0,0,0,0,0}, {0,0,0,0,0,0,0,0}},
                      {{0,0,0,0,0,0,0,0}, {0,0,0,0,0,0,0,0},
                       {0,0,0,0,0,0,0,0}, {0,0,0,0,0,0,0,0}}};
    v8i accH[2][4] = {{{0,0,0,0,0,0,0,0}, {0,0,0,0,0,0,0,0},
                       {0,0,0,0,0,0,0,0}, {0,0,0,0,0,0,0,0}},
                      {{0,0,0,0,0,0,0,0}, {0,0,0,0,0,0,0,0},
                       {0,0,0,0,0,0,0,0}, {0,0,0,0,0,0,0,0}}};

    {
        v8i ax0 = load_a8(&xq[(wave * 32 + lm) * 64 + hi * 8]);
        v8i ax1 = load_a8(&xq[(wave * 32 + 16 + lm) * 64 + hi * 8]);
#pragma unroll
        for (int g = 0; g < 4; ++g) {
            const int col = g * HDIM + hc0 + lm;
            v8i bv = load_b8(wih_p + ((size_t)nb * GATES + col) * 64 + hi * 32);
            accX[0][g] = __builtin_amdgcn_wmma_i32_16x16x64_iu8(
                true, ax0, true, bv, accX[0][g], false, false);
            accX[1][g] = __builtin_amdgcn_wmma_i32_16x16x64_iu8(
                true, ax1, true, bv, accX[1][g], false, false);
        }
    }
    const size_t hrow0 = ((size_t)nb * BATCH + r0 + wave * 32 + lm) * HDIM;
    const size_t hrow1 = hrow0 + (size_t)16 * HDIM;
#pragma unroll
    for (int kt = 0; kt < KT_HH; ++kt) {
        v8i av0 = load_a8(h_in + hrow0 + kt * 64 + hi * 8);
        v8i av1 = load_a8(h_in + hrow1 + kt * 64 + hi * 8);
        if (kt + 1 < KT_HH) {
            __builtin_prefetch(
                whh_p + (((size_t)nb * KT_HH + kt + 1) * GATES + hc0 + lm) * 64,
                0, 3);
        }
#pragma unroll
        for (int g = 0; g < 4; ++g) {
            const int col = g * HDIM + hc0 + lm;
            v8i bv = load_b8(
                whh_p + (((size_t)nb * KT_HH + kt) * GATES + col) * 64 + hi * 32);
            accH[0][g] = __builtin_amdgcn_wmma_i32_16x16x64_iu8(
                true, av0, true, bv, accH[0][g], false, false);
            accH[1][g] = __builtin_amdgcn_wmma_i32_16x16x64_iu8(
                true, av1, true, bv, accH[1][g], false, false);
        }
    }

    const float s1 = a1 * (1.f / 16384.f);
    const float s2 = a11 * (1.f / 16384.f);
    float b1q[4], b2q[4];
#pragma unroll
    for (int g = 0; g < 4; ++g) {
        const int col = g * HDIM + hc0 + lm;
        b1q[g] = quant8_r1(bih[(size_t)nb * GATES + col]);
        b2q[g] = quant8_r1(bhh[(size_t)nb * GATES + col]);
    }
#pragma unroll
    for (int mi = 0; mi < 2; ++mi) {
#pragma unroll
        for (int e = 0; e < 8; ++e) {
            const int row = r0 + wave * 32 + mi * 16 + hi * 8 + e;
            const int hcol = hc0 + lm;
            float gv[4];
#pragma unroll
            for (int g = 0; g < 4; ++g) {
                float p1 = (float)accX[mi][g][e] * s1 + b1q[g];
                float p2 = (float)accH[mi][g][e] * s2 + b2q[g];
                float pre = pact_q8(clamp_pm(p1, a12), a12) +
                            pact_q8(clamp_pm(p2, a13), a13);
                gv[g] = pact_q8(clamp_pm(pre, a14), a14);
            }
            float ig = pact_q8(clamp_pm(sigmoidf(gv[0]), a4), a4);
            float jg = pact_q8(clamp_pm(tanhf(gv[1]), a5), a5);
            float fg = pact_q8(clamp_pm(sigmoidf(gv[2]), a3), a3);
            float og = pact_q8(clamp_pm(sigmoidf(gv[3]), a6), a6);

            const size_t cidx = ((size_t)nb * BATCH + row) * HDIM + hcol;
            float cx = c[cidx];
            float gc = pact_q8(clamp_pm(cx * fg, a7), a7);
            float ai = pact_q8(clamp_pm(ig * jg, a8), a8);
            float nc = pact_q8(clamp_pm(gc + ai, a9), a9);
            float ac = pact_q8(clamp_pm(tanhf(nc), a10), a10);
            float nhv = clamp_pm(ac * og, a11);
            c[cidx] = nc;
            h_out[cidx] = (signed char)rintf(
                fminf(fmaxf(nhv * (128.f / a11), -127.f), 127.f));
        }
    }
}

__global__ __launch_bounds__(256) void fc_kernel(
    const signed char* __restrict__ hT,
    const float* __restrict__ fcw,
    const float* __restrict__ fcb,
    const float* __restrict__ fca,
    const float* __restrict__ a_cell,
    float* __restrict__ outp) {
    int idx = blockIdx.x * 256 + threadIdx.x;
    if (idx >= BATCH * 12) return;
    const int b = idx / 12;
    const int rem = idx % 12;
    const int nb = rem / 3;
    const int o = rem % 3;
    const float a11 = a_cell[10 * NBLK + nb];
    const float fa1 = fca[0 * NBLK + nb];
    const float fa2 = fca[1 * NBLK + nb];
    const float sh = a11 * 0.0078125f;
    float acc = 0.f;
    const signed char* hrow = hT + ((size_t)nb * BATCH + b) * HDIM;
    for (int k = 0; k < HDIM; ++k) {
        float hv = (float)hrow[k] * sh;
        float xv = pact_q8(clamp_pm(hv, fa1), fa1);
        float wq = quant8_r1(fminf(fmaxf(fcw[((size_t)nb * HDIM + k) * 3 + o],
                                         -1.f), 1.f));
        acc = fmaf(xv, wq, acc);
    }
    acc += quant8_r1(fcb[nb * 3 + o]);
    outp[(size_t)b * 12 + rem] = pact_q8(clamp_pm(acc, fa2), fa2);
}

extern "C" void kernel_launch(void* const* d_in, const int* in_sizes, int n_in,
                              void* d_out, int out_size, void* d_ws, size_t ws_size,
                              hipStream_t stream) {
    const float* inputs = (const float*)d_in[0];
    const float* w_ih   = (const float*)d_in[1];
    const float* w_hh   = (const float*)d_in[2];
    const float* b_ih   = (const float*)d_in[3];
    const float* b_hh   = (const float*)d_in[4];
    const float* a_cell = (const float*)d_in[5];
    const float* fc_w   = (const float*)d_in[6];
    const float* fc_b   = (const float*)d_in[7];
    const float* fc_a   = (const float*)d_in[8];
    float* outp = (float*)d_out;

    char* ws = (char*)d_ws;
    signed char* whh_p = (signed char*)(ws);
    signed char* wih_p = (signed char*)(ws + 4194304);
    signed char* h0    = (signed char*)(ws + 4718592);
    signed char* h1    = (signed char*)(ws + 6815744);
    float*       cbuf  = (float*)(ws + 8912896);
    (void)in_sizes; (void)n_in; (void)out_size; (void)ws_size;

    pack_whh_kernel<<<16384, 256, 0, stream>>>(w_hh, whh_p);
    pack_wih_kernel<<<2048, 256, 0, stream>>>(w_ih, wih_p);
    zero_state_kernel<<<8192, 256, 0, stream>>>(h0, cbuf);

    for (int t = 0; t < T_STEPS; ++t) {
        const float* xt = inputs + (size_t)t * BATCH * INDIM;
        const signed char* hin = (t & 1) ? h1 : h0;
        signed char* hout = (t & 1) ? h0 : h1;
        lstm_step_kernel<<<dim3(HDIM / 16, BATCH / 256, NBLK), 256, 0, stream>>>(
            xt, hin, hout, cbuf, wih_p, whh_p, b_ih, b_hh, a_cell);
    }
    fc_kernel<<<(BATCH * 12 + 255) / 256, 256, 0, stream>>>(
        h0, fc_w, fc_b, fc_a, a_cell, outp);
}
